// GATNet_10368051052624
// MI455X (gfx1250) — matrix operand order repaired and checked by reading, not compiled, not run on hardware
//
#include <hip/hip_runtime.h>
#include <hip/hip_bf16.h>
#include <stdint.h>

// ---------------------------------------------------------------------------
// Types for CDNA5 WMMA (wave32): V_WMMA_F32_16X16X32_BF16
// ---------------------------------------------------------------------------
typedef __bf16 bf16;
typedef __attribute__((ext_vector_type(16))) __bf16 v16bf;
typedef __attribute__((ext_vector_type(8)))  float  v8f;

union Frag16 { v16bf v; uint32_t u[8]; uint4 q[2]; };
union FragC  { v8f   v; float    f[8]; };

__device__ __forceinline__ void atomicMaxF(float* addr, float val) {
  // Sign-aware float max via integer atomics (works for mixed signs, -inf init).
  if (val >= 0.0f) atomicMax((int*)addr, __float_as_int(val));
  else             atomicMin((unsigned int*)addr, __float_as_uint(val));
}

// ---------------------------------------------------------------------------
// W pre-transform: W[k][n] (fp32, K=128 rows) -> WT_hi/WT_lo[n][k] (bf16 split)
// ---------------------------------------------------------------------------
__global__ void wprep_kernel(const float* __restrict__ W,
                             bf16* __restrict__ WTh, bf16* __restrict__ WTl, int M) {
  int idx = blockIdx.x * blockDim.x + threadIdx.x;
  if (idx >= 128 * M) return;
  int k = idx / M, n = idx % M;
  float v = W[idx];
  bf16 hi = (bf16)v;
  WTh[(size_t)n * 128 + k] = hi;
  WTl[(size_t)n * 128 + k] = (bf16)(v - (float)hi);
}

__global__ void init_kernel(float* __restrict__ OUT, long long nOut,
                            float* __restrict__ MX, float* __restrict__ DEN,
                            long long nMH) {
  long long i = (long long)blockIdx.x * blockDim.x + threadIdx.x;
  if (i < nOut) OUT[i] = 0.0f;
  if (i < nMH) { MX[i] = -__builtin_inff(); DEN[i] = 0.0f; }
}

// ---------------------------------------------------------------------------
// GEMM: H[N,MOUT] = X[N,128] @ W[128,MOUT] via bf16 split-product WMMA
// (hi*hi + hi*lo + lo*hi  ->  ~fp32 precision, fp32 accumulate)
// Block: 128 threads = 4 wave32; each wave owns a 16-row M tile.
// ---------------------------------------------------------------------------
template<int MOUT>
__global__ __launch_bounds__(128)
void gemm_wmma(const float* __restrict__ X, const bf16* __restrict__ WTh,
               const bf16* __restrict__ WTl, float* __restrict__ Hout, int N) {
  // wave-private A staging: [wave][hi/lo][m=16][k=32 (+8 pad)]
  __shared__ __align__(16) bf16 sA[4][2][16][40];

  const int lane = threadIdx.x & 31;
  const int wave = threadIdx.x >> 5;
  const int m = lane & 15;     // fragment row (A) / col (B,C,D)
  const int g = lane >> 4;     // lane half
  const int rowBase = (blockIdx.x * 4 + wave) * 16;

  FragC acc[MOUT / 16];
  #pragma unroll
  for (int t = 0; t < MOUT / 16; ++t)
    #pragma unroll
    for (int r = 0; r < 8; ++r) acc[t].f[r] = 0.0f;

  for (int kc = 0; kc < 4; ++kc) {   // K = 128 = 4 x 32
    // ---- stage A chunk: lane loads 16 fp32, splits into bf16 hi/lo ----
    {
      const int k0 = kc * 32 + g * 16;
      const int gm = rowBase + m;
      float vals[16];
      if (gm < N) {
        const float4* p = (const float4*)(X + (size_t)gm * 128 + k0);
        #pragma unroll
        for (int qq = 0; qq < 4; ++qq) {
          float4 f = p[qq];
          vals[qq*4+0] = f.x; vals[qq*4+1] = f.y;
          vals[qq*4+2] = f.z; vals[qq*4+3] = f.w;
        }
      } else {
        #pragma unroll
        for (int j = 0; j < 16; ++j) vals[j] = 0.0f;
      }
      #pragma unroll
      for (int j = 0; j < 16; ++j) {
        float fv = vals[j];
        bf16 hi = (bf16)fv;
        bf16 lo = (bf16)(fv - (float)hi);
        sA[wave][0][m][g * 16 + j] = hi;
        sA[wave][1][m][g * 16 + j] = lo;
      }
    }
    __syncthreads();

    // ---- A fragments (contiguous K, matching B): lane m, K pairs:
    // K = g*16 + 2v  => two b128 loads ----
    Frag16 aH, aL;
    #pragma unroll
    for (int h2 = 0; h2 < 2; ++h2) {
      int kk = g * 16 + h2 * 8;
      aH.q[h2] = *(const uint4*)&sA[wave][0][m][kk];
      aL.q[h2] = *(const uint4*)&sA[wave][1][m][kk];
    }

    // ---- sweep all N tiles, B frags straight from L2-resident WT ----
    #pragma unroll
    for (int t = 0; t < MOUT / 16; ++t) {
      const int n = t * 16 + m;
      // B layout: lane col n, K = g*16 + 2v (+1) -> 16 contiguous bf16
      const size_t wb = (size_t)n * 128 + kc * 32 + g * 16;
      Frag16 bH, bL;
      bH.q[0] = *(const uint4*)(WTh + wb);
      bH.q[1] = *(const uint4*)(WTh + wb + 8);
      bL.q[0] = *(const uint4*)(WTl + wb);
      bL.q[1] = *(const uint4*)(WTl + wb + 8);
      acc[t].v = __builtin_amdgcn_wmma_f32_16x16x32_bf16(
          false, aH.v, false, bH.v, (short)0, acc[t].v, false, false);
      acc[t].v = __builtin_amdgcn_wmma_f32_16x16x32_bf16(
          false, aH.v, false, bL.v, (short)0, acc[t].v, false, false);
      acc[t].v = __builtin_amdgcn_wmma_f32_16x16x32_bf16(
          false, aL.v, false, bH.v, (short)0, acc[t].v, false, false);
    }
    __syncthreads();
  }

  // ---- D layout: VGPR r -> row (r + 8*g), col n = t*16 + m ----
  #pragma unroll
  for (int t = 0; t < MOUT / 16; ++t) {
    const int n = t * 16 + m;
    #pragma unroll
    for (int r = 0; r < 8; ++r) {
      const int gm = rowBase + r + 8 * g;
      if (gm < N) Hout[(size_t)gm * MOUT + n] = acc[t].f[r];
    }
  }
}

// ---------------------------------------------------------------------------
// Attention scalars: AL/AD[n,h] = dot(h[n,h,:], a_src/a_dst[h,:])
// ---------------------------------------------------------------------------
__global__ void attn_kernel(const float* __restrict__ Hf, const float* __restrict__ aS,
                            const float* __restrict__ aD, float* __restrict__ AL,
                            float* __restrict__ ADv, int N, int C, int MOUT) {
  int idx = blockIdx.x * blockDim.x + threadIdx.x;
  if (idx >= N * 4) return;
  int n = idx >> 2, h = idx & 3;
  const float* hp = Hf + (size_t)n * MOUT + h * C;
  const float* as = aS + h * C;
  const float* ad = aD + h * C;
  float s = 0.0f, d = 0.0f;
  for (int c = 0; c < C; c += 4) {
    float4 hv = *(const float4*)(hp + c);
    float4 sv = *(const float4*)(as + c);
    float4 dv = *(const float4*)(ad + c);
    s += hv.x * sv.x + hv.y * sv.y + hv.z * sv.z + hv.w * sv.w;
    d += hv.x * dv.x + hv.y * dv.y + hv.z * dv.z + hv.w * dv.w;
  }
  AL[idx] = s; ADv[idx] = d;
}

// ---------------------------------------------------------------------------
// Edge phase. Edge list = edge_index[2,E] (int64) + implicit self loops.
// ---------------------------------------------------------------------------
__global__ void edge_logits_max_kernel(const long long* __restrict__ ei, int E, int nE,
                                       const float* __restrict__ AL,
                                       const float* __restrict__ ADv,
                                       float* __restrict__ LOG, float* __restrict__ MX) {
  int e = blockIdx.x * blockDim.x + threadIdx.x;
  if (e >= nE) return;
  long long s, d;
  if (e < E) { s = ei[e]; d = ei[(size_t)E + e]; } else { s = d = (long long)(e - E); }
  float4 ls = *(const float4*)(AL + s * 4);
  float4 ld = *(const float4*)(ADv + d * 4);
  float lg[4] = { ls.x + ld.x, ls.y + ld.y, ls.z + ld.z, ls.w + ld.w };
  #pragma unroll
  for (int h = 0; h < 4; ++h) {
    float v = lg[h];
    v = (v >= 0.0f) ? v : 0.2f * v;     // leaky_relu, slope 0.2
    lg[h] = v;
    atomicMaxF(MX + d * 4 + h, v);
  }
  float4 st = { lg[0], lg[1], lg[2], lg[3] };
  *(float4*)(LOG + (size_t)e * 4) = st;
}

__global__ void edge_exp_kernel(const long long* __restrict__ ei, int E, int nE,
                                float* __restrict__ LOG, const float* __restrict__ MX,
                                float* __restrict__ DEN) {
  int e = blockIdx.x * blockDim.x + threadIdx.x;
  if (e >= nE) return;
  long long d;
  if (e < E) d = ei[(size_t)E + e]; else d = (long long)(e - E);
  float4 l = *(const float4*)(LOG + (size_t)e * 4);
  float4 mm = *(const float4*)(MX + d * 4);
  float m0 = isfinite(mm.x) ? mm.x : 0.0f;
  float m1 = isfinite(mm.y) ? mm.y : 0.0f;
  float m2 = isfinite(mm.z) ? mm.z : 0.0f;
  float m3 = isfinite(mm.w) ? mm.w : 0.0f;
  float ex[4] = { expf(l.x - m0), expf(l.y - m1), expf(l.z - m2), expf(l.w - m3) };
  float4 st = { ex[0], ex[1], ex[2], ex[3] };
  *(float4*)(LOG + (size_t)e * 4) = st;     // overwrite logits with exp values
  #pragma unroll
  for (int h = 0; h < 4; ++h) atomicAdd(DEN + d * 4 + h, ex[h]);
}

// One wave32 per edge; lanes cover MOUT features as float4; L2-resident atomics.
__global__ __launch_bounds__(256)
void edge_scatter_kernel(const long long* __restrict__ ei, int E, int nE,
                         const float* __restrict__ EV, const float* __restrict__ DEN,
                         const float* __restrict__ Hf, float* __restrict__ OUT,
                         int MOUT, int logC) {
  int warp = blockIdx.x * 8 + (threadIdx.x >> 5);
  int lane = threadIdx.x & 31;
  if (warp >= nE) return;
  long long s, d;
  if (warp < E) { s = ei[warp]; d = ei[(size_t)E + warp]; } else { s = d = (long long)(warp - E); }
  const float* hrow = Hf + (size_t)s * MOUT;
  float* orow = OUT + (size_t)d * MOUT;
  for (int f0 = lane * 4; f0 < MOUT; f0 += 128) {
    int h = f0 >> logC;                               // head of this float4 (4 | C)
    float alpha = EV[(size_t)warp * 4 + h] / DEN[d * 4 + h];
    float4 hv = *(const float4*)(hrow + f0);
    atomicAdd(orow + f0 + 0, alpha * hv.x);
    atomicAdd(orow + f0 + 1, alpha * hv.y);
    atomicAdd(orow + f0 + 2, alpha * hv.z);
    atomicAdd(orow + f0 + 3, alpha * hv.w);
  }
}

// ---------------------------------------------------------------------------
// Epilogues
// ---------------------------------------------------------------------------
__global__ void elu_bias_kernel(const float* __restrict__ OUT, const float* __restrict__ b,
                                float* __restrict__ Xn, long long n) {
  long long i = (long long)blockIdx.x * blockDim.x + threadIdx.x;
  if (i >= n) return;
  float v = OUT[i] + b[i & 127];
  Xn[i] = (v > 0.0f) ? v : expm1f(v);
}

__global__ void mean_bias_kernel(const float* __restrict__ OUT, const float* __restrict__ b3,
                                 float* __restrict__ Y, int N) {
  int idx = blockIdx.x * blockDim.x + threadIdx.x;
  if (idx >= N * 64) return;
  int n = idx >> 6, c = idx & 63;
  const float* o = OUT + (size_t)n * 256 + c;
  Y[idx] = 0.25f * (o[0] + o[64] + o[128] + o[192]) + b3[c];
}

// ---------------------------------------------------------------------------
// Host launcher
// ---------------------------------------------------------------------------
extern "C" void kernel_launch(void* const* d_in, const int* in_sizes, int n_in,
                              void* d_out, int out_size, void* d_ws, size_t ws_size,
                              hipStream_t stream) {
  (void)n_in; (void)out_size; (void)ws_size;
  const float*     x   = (const float*)d_in[0];
  const long long* ei  = (const long long*)d_in[1];     // int64 edge_index [2,E]
  const float* W1  = (const float*)d_in[2];
  const float* aS1 = (const float*)d_in[3];
  const float* aD1 = (const float*)d_in[4];
  const float* b1  = (const float*)d_in[5];
  const float* W2  = (const float*)d_in[6];
  const float* aS2 = (const float*)d_in[7];
  const float* aD2 = (const float*)d_in[8];
  const float* b2  = (const float*)d_in[9];
  const float* W3  = (const float*)d_in[10];
  const float* aS3 = (const float*)d_in[11];
  const float* aD3 = (const float*)d_in[12];
  const float* b3  = (const float*)d_in[13];

  const int N  = in_sizes[0] / 128;
  const int E  = in_sizes[1] / 2;
  const int nE = E + N;                                  // + self loops

  char* ws = (char*)d_ws;
  size_t off = 0;
  auto alloc = [&](size_t bytes) -> char* {
    char* p = ws + off;
    off += (bytes + 255) & ~(size_t)255;
    return p;
  };
  bf16*  WTh = (bf16*)alloc((size_t)256 * 128 * sizeof(bf16));
  bf16*  WTl = (bf16*)alloc((size_t)256 * 128 * sizeof(bf16));
  float* Hf  = (float*)alloc((size_t)N * 256 * 4);
  float* OUT = (float*)alloc((size_t)N * 256 * 4);
  float* Xb  = (float*)alloc((size_t)N * 128 * 4);
  float* AL  = (float*)alloc((size_t)N * 16);
  float* ADv = (float*)alloc((size_t)N * 16);
  float* MX  = (float*)alloc((size_t)N * 16);
  float* DEN = (float*)alloc((size_t)N * 16);
  float* LOG = (float*)alloc((size_t)nE * 16);

  auto layer = [&](const float* Xin, const float* W, const float* aS, const float* aD,
                   int MOUT, int C, int logC) {
    int wn = 128 * MOUT;
    wprep_kernel<<<(wn + 255) / 256, 256, 0, stream>>>(W, WTh, WTl, MOUT);
    long long nOut = (long long)N * MOUT;
    init_kernel<<<(unsigned)((nOut + 255) / 256), 256, 0, stream>>>(
        OUT, nOut, MX, DEN, (long long)N * 4);
    if (MOUT == 128)
      gemm_wmma<128><<<(N + 63) / 64, 128, 0, stream>>>(Xin, WTh, WTl, Hf, N);
    else
      gemm_wmma<256><<<(N + 63) / 64, 128, 0, stream>>>(Xin, WTh, WTl, Hf, N);
    attn_kernel<<<(N * 4 + 255) / 256, 256, 0, stream>>>(Hf, aS, aD, AL, ADv, N, C, MOUT);
    edge_logits_max_kernel<<<(nE + 255) / 256, 256, 0, stream>>>(ei, E, nE, AL, ADv, LOG, MX);
    edge_exp_kernel<<<(nE + 255) / 256, 256, 0, stream>>>(ei, E, nE, LOG, MX, DEN);
    edge_scatter_kernel<<<(nE + 7) / 8, 256, 0, stream>>>(ei, E, nE, LOG, DEN, Hf, OUT, MOUT, logC);
  };

  layer(x, W1, aS1, aD1, 128, 32, 5);
  elu_bias_kernel<<<(unsigned)(((long long)N * 128 + 255) / 256), 256, 0, stream>>>(
      OUT, b1, Xb, (long long)N * 128);
  layer(Xb, W2, aS2, aD2, 128, 32, 5);
  elu_bias_kernel<<<(unsigned)(((long long)N * 128 + 255) / 256), 256, 0, stream>>>(
      OUT, b2, Xb, (long long)N * 128);
  layer(Xb, W3, aS3, aD3, 256, 64, 6);
  mean_bias_kernel<<<(N * 64 + 255) / 256, 256, 0, stream>>>(OUT, b3, (float*)d_out, N);
}
